// Multi_Head_Attention_82557861364136
// MI455X (gfx1250) — matrix operand order repaired and checked by reading, not compiled, not run on hardware
//
#include <hip/hip_runtime.h>
#include <hip/hip_bf16.h>

// ---------------------------------------------------------------------------
// Multi-head attention (B=1, C=256, H=W=48, 4 heads) for gfx1250 (MI455X).
// Flash-attention pipeline: v_wmma_f32_16x16x32_f16 for all matmuls,
// TENSOR_LOAD_TO_LDS (TDM) for K/V chunk staging when available.
// ---------------------------------------------------------------------------

typedef _Float16 half_t;
typedef __attribute__((ext_vector_type(16))) _Float16     v16h;
typedef __attribute__((ext_vector_type(8)))  float        v8f;
typedef __attribute__((ext_vector_type(4)))  unsigned int v4u;
typedef __attribute__((ext_vector_type(8)))  int          v8i;
typedef __attribute__((ext_vector_type(4)))  int          v4i;

union FragH { v16h h; unsigned int u[8]; };

__device__ __forceinline__ v8f wmma_f16(v16h a, v16h b, v8f c) {
  return __builtin_amdgcn_wmma_f32_16x16x32_f16(
      false, a, false, b, (short)0, c, false, false);
}

// A-operand (16x32 f16), src row-major [row][k], ld in halfs (even).
// lane m (=lane&15), half=lane>>4; VGPR v holds K pair
//   k = kbase + (v>>2)*16 + half*8 + (v&3)*2
__device__ __forceinline__ v16h load_a_frag(const half_t* src, int ld,
                                            int rowbase, int kbase) {
  const int lane = threadIdx.x & 31;
  const int hid  = lane >> 4;
  const int m    = lane & 15;
  const unsigned int* p32 =
      reinterpret_cast<const unsigned int*>(src + (rowbase + m) * ld);
  FragH f;
#pragma unroll
  for (int v = 0; v < 8; ++v) {
    int k = kbase + ((v >> 2) << 4) + (hid << 3) + ((v & 3) << 1);
    f.u[v] = p32[k >> 1];
  }
  return f.h;
}

// B-operand (32x16 f16) where B[k][n] = src[n*ld + k] (K contiguous per row).
// lane n (=lane&15) holds column nbase+n; half=lane>>4;
//   VGPR v holds K pair k = kbase + (v>>2)*16 + half*8 + (v&3)*2
__device__ __forceinline__ v16h load_b_fragT(const half_t* src, int ld,
                                             int nbase, int kbase) {
  const int lane = threadIdx.x & 31;
  const int hid  = lane >> 4;
  const int n    = nbase + (lane & 15);
  const unsigned int* p32 =
      reinterpret_cast<const unsigned int*>(src + n * ld);
  FragH f;
#pragma unroll
  for (int v = 0; v < 8; ++v) {
    int k = kbase + ((v >> 2) << 4) + (hid << 3) + ((v & 3) << 1);
    f.u[v] = p32[k >> 1];
  }
  return f.h;
}

// ---------------------------------------------------------------------------
// Tensor Data Mover support (6-arg builtin: g0, g1, g2, g3, extra, cpol).
// Falls back to manual global->LDS staging if the builtin is unavailable.
// ---------------------------------------------------------------------------
#if __has_builtin(__builtin_amdgcn_tensor_load_to_lds) && \
    __has_builtin(__builtin_amdgcn_s_wait_tensorcnt)
#define MHA_USE_TDM 1
#else
#define MHA_USE_TDM 0
#endif

#if MHA_USE_TDM
__device__ __forceinline__ unsigned lds_addr_u32(void* p) {
  return (unsigned)(unsigned long long)
      (__attribute__((address_space(3))) void*)p;
}

// 2D tile DMA: tile_d0 contiguous f16 elements x tile_d1 rows, global row
// stride stride_d0 (elements), into LDS with hardware padding of 1 DWORD
// every (2<<pad_code) DWORDs (pad_code 4 -> 66-half rows, 3 -> 34-half rows).
__device__ __forceinline__ void tdm_load_2d(
    void* lds_dst, const void* gsrc,
    unsigned tensor_d0, unsigned tensor_d1,
    unsigned tile_d0, unsigned tile_d1,
    unsigned stride_d0, unsigned pad_code) {
  const unsigned long long ga = (unsigned long long)gsrc;
  v4u g0;
  g0[0] = 1u;                                        // count=1 (valid D#)
  g0[1] = lds_addr_u32(lds_dst);                     // lds_addr
  g0[2] = (unsigned)(ga & 0xffffffffu);              // global_addr[31:0]
  g0[3] = (unsigned)((ga >> 32) & 0x1ffffffu)        // global_addr[56:32]
        | (2u << 30);                                // type=2 ("image")
  v8i g1;
  g1[0] = (int)((1u << 16)                           // data_size = 2 bytes
              | (1u << 20)                           // pad_enable
              | (pad_code << 22));                   // pad_interval (amount=1DW)
  g1[1] = (int)(tensor_d0 << 16);                    // tensor_dim0[15:0]
  g1[2] = (int)((tensor_d0 >> 16) | (tensor_d1 << 16));
  g1[3] = (int)((tensor_d1 >> 16) | (tile_d0 << 16));
  g1[4] = (int)(tile_d1 & 0xffffu);                  // tile_dim1 (tile_dim2=0)
  g1[5] = (int)stride_d0;                            // tensor_dim0_stride lo
  g1[6] = 0;
  g1[7] = 0;
  const v4i gz4 = {0, 0, 0, 0};
  const v8i gz8 = {0, 0, 0, 0, 0, 0, 0, 0};
  __builtin_amdgcn_tensor_load_to_lds(g0, g1, gz4, gz4, gz8, 0);
}
#endif

// ---------------------------------------------------------------------------
// Kernel 1: pack f32 -> f16. Xh[p][c] = x[c][p]; weights copied as-is.
// ---------------------------------------------------------------------------
__global__ __launch_bounds__(256) void mha_pack(
    const float* __restrict__ x,
    const float* __restrict__ Wq, const float* __restrict__ Wk,
    const float* __restrict__ Wv, const float* __restrict__ Wo,
    half_t* __restrict__ Xh,
    half_t* __restrict__ Wqh, half_t* __restrict__ Wkh,
    half_t* __restrict__ Wvh, half_t* __restrict__ Woh) {
  const int bid = blockIdx.x;
  const int t   = threadIdx.x;
  if (bid < 2304) {
    Xh[bid * 256 + t] = (half_t)x[t * 2304 + bid];
  } else {
    const int w = (bid - 2304) >> 8;
    const int i = ((bid - 2304) & 255) * 256 + t;
    const float* src = (w == 0) ? Wq : (w == 1) ? Wk : (w == 2) ? Wv : Wo;
    half_t*      dst = (w == 0) ? Wqh : (w == 1) ? Wkh : (w == 2) ? Wvh : Woh;
    dst[i] = (half_t)src[i];
  }
}

// ---------------------------------------------------------------------------
// Kernel 2: QKV projection. Q scaled by 1/8 and stored [9216][64]-compatible;
// V stored TRANSPOSED as Vt[64][9216] so attention's PV B-operand tiles are
// plain strided 2D tiles (TDM-friendly, no in-kernel transpose needed).
// ---------------------------------------------------------------------------
__global__ __launch_bounds__(256) void mha_qkv(
    const half_t* __restrict__ Xh,
    const half_t* __restrict__ Wqh, const half_t* __restrict__ Wkh,
    const half_t* __restrict__ Wvh,
    const float* __restrict__ bq, const float* __restrict__ bk,
    const float* __restrict__ bv,
    half_t* __restrict__ Qh, half_t* __restrict__ Kh, half_t* __restrict__ Vt) {
  const int which = blockIdx.y;
  const half_t* W = (which == 0) ? Wqh : (which == 1) ? Wkh : Wvh;
  const float*  b = (which == 0) ? bq  : (which == 1) ? bk  : bv;
  const float scale = (which == 0) ? 0.125f : 1.0f;

  const int rowbase = blockIdx.x * 16;
  const int wave = threadIdx.x >> 5;
  const int lane = threadIdx.x & 31;
  const int hid  = lane >> 4;
  const int nl   = lane & 15;
  const int nb   = wave * 32;

  v8f acc0 = {}, acc1 = {};
#pragma unroll
  for (int kc = 0; kc < 8; ++kc) {
    v16h a  = load_a_frag(Xh, 256, rowbase, kc * 32);
    v16h b0 = load_b_fragT(W, 256, nb,      kc * 32);
    v16h b1 = load_b_fragT(W, 256, nb + 16, kc * 32);
    acc0 = wmma_f16(a, b0, acc0);
    acc1 = wmma_f16(a, b1, acc1);
  }
  const int   c0 = nb + nl, c1 = nb + 16 + nl;
  const float bias0 = b[c0], bias1 = b[c1];

  if (which == 2) {
    // Vt[d][n]: d = col&63, n = pixel*4 + (col>>6)
#pragma unroll
    for (int r = 0; r < 8; ++r) {
      const int grow = rowbase + r + hid * 8;
      Vt[(c0 & 63) * 9216 + grow * 4 + (c0 >> 6)] = (half_t)(acc0[r] + bias0);
      Vt[(c1 & 63) * 9216 + grow * 4 + (c1 >> 6)] = (half_t)(acc1[r] + bias1);
    }
  } else {
    half_t* O = (which == 0) ? Qh : Kh;
#pragma unroll
    for (int r = 0; r < 8; ++r) {
      const int grow = rowbase + r + hid * 8;
      O[grow * 256 + c0] = (half_t)((acc0[r] + bias0) * scale);
      O[grow * 256 + c1] = (half_t)((acc1[r] + bias1) * scale);
    }
  }
}

// ---------------------------------------------------------------------------
// Kernel 3: flash attention. Qf/Kf = [9216][64] f16; Vt = [64][9216] f16.
// 128 threads = 4 waves, 16 query rows each; 32-key chunks staged via TDM.
// ---------------------------------------------------------------------------
__global__ __launch_bounds__(128) void mha_attn(
    const half_t* __restrict__ Qf, const half_t* __restrict__ Kf,
    const half_t* __restrict__ Vt, half_t* __restrict__ Oh) {
  __shared__ half_t Kl[32 * 66];        // K chunk   [key][d],  ld=66
  __shared__ half_t Vl[64 * 34];        // Vt chunk  [d][key],  ld=34
  __shared__ half_t Pl[4][16 * 34];     // per-wave P tile [m][key], ld=34

  const int wave = threadIdx.x >> 5;
  const int lane = threadIdx.x & 31;
  const int hid  = lane >> 4;
  const int nl   = lane & 15;
  const int qrow = blockIdx.x * 64 + wave * 16;

  const v16h qa0 = load_a_frag(Qf, 64, qrow, 0);
  const v16h qa1 = load_a_frag(Qf, 64, qrow, 32);

  float mrow[8], lrow[8];
#pragma unroll
  for (int r = 0; r < 8; ++r) { mrow[r] = -1e30f; lrow[r] = 0.0f; }
  v8f o0 = {}, o1 = {}, o2 = {}, o3 = {};

#if !MHA_USE_TDM
  const int keyr  = threadIdx.x >> 2;        // K staging: 0..31
  const int dbase = (threadIdx.x & 3) * 16;
  const int vrow  = threadIdx.x >> 1;        // V staging: 0..63
  const int vbase = (threadIdx.x & 1) * 16;
#endif

  for (int c = 0; c < 9216; c += 32) {
    __syncthreads();   // previous chunk's LDS reads complete
#if MHA_USE_TDM
    if (wave == 0) {
      // K rows [c, c+32): 64 halfs/row, row stride 64, LDS rows 66 halfs
      tdm_load_2d(Kl, Kf + c * 64, /*t_d0=*/64, /*t_d1=*/9216,
                  /*tile_d0=*/64, /*tile_d1=*/32, /*stride=*/64, /*pad=*/4);
      __builtin_amdgcn_s_wait_tensorcnt(0);
    } else if (wave == 1) {
      // Vt cols [c, c+32) of all 64 rows: 32 halfs/row, stride 9216 -> ld 34
      tdm_load_2d(Vl, Vt + c, /*t_d0=*/9216, /*t_d1=*/64,
                  /*tile_d0=*/32, /*tile_d1=*/64, /*stride=*/9216, /*pad=*/3);
      __builtin_amdgcn_s_wait_tensorcnt(0);
    }
#else
    {
      const uint4* ks = reinterpret_cast<const uint4*>(Kf + (c + keyr) * 64 + dbase);
      uint4 kA = ks[0], kB = ks[1];
      unsigned int* kd = reinterpret_cast<unsigned int*>(&Kl[keyr * 66 + dbase]);
      kd[0] = kA.x; kd[1] = kA.y; kd[2] = kA.z; kd[3] = kA.w;
      kd[4] = kB.x; kd[5] = kB.y; kd[6] = kB.z; kd[7] = kB.w;
      const uint4* vs = reinterpret_cast<const uint4*>(Vt + vrow * 9216 + c + vbase);
      uint4 vA = vs[0], vB = vs[1];
      unsigned int* vd = reinterpret_cast<unsigned int*>(&Vl[vrow * 34 + vbase]);
      vd[0] = vA.x; vd[1] = vA.y; vd[2] = vA.z; vd[3] = vA.w;
      vd[4] = vB.x; vd[5] = vB.y; vd[6] = vB.z; vd[7] = vB.w;
    }
#endif
    __syncthreads();

    // ---- S tile = Q(16x64) @ K^T(64x32) ----
    v8f s0 = {}, s1 = {};
    s0 = wmma_f16(qa0, load_b_fragT(Kl, 66, 0, 0),  s0);
    s0 = wmma_f16(qa1, load_b_fragT(Kl, 66, 0, 32), s0);
    s1 = wmma_f16(qa0, load_b_fragT(Kl, 66, 16, 0),  s1);
    s1 = wmma_f16(qa1, load_b_fragT(Kl, 66, 16, 32), s1);

    // ---- online softmax ----
#pragma unroll
    for (int r = 0; r < 8; ++r) {
      float mx = fmaxf(s0[r], s1[r]);
      mx = fmaxf(mx, __shfl_xor(mx, 8, 32));
      mx = fmaxf(mx, __shfl_xor(mx, 4, 32));
      mx = fmaxf(mx, __shfl_xor(mx, 2, 32));
      mx = fmaxf(mx, __shfl_xor(mx, 1, 32));
      const float mnew  = fmaxf(mrow[r], mx);
      const float alpha = __expf(mrow[r] - mnew);
      const float p0 = __expf(s0[r] - mnew);
      const float p1 = __expf(s1[r] - mnew);
      float ps = p0 + p1;
      ps += __shfl_xor(ps, 8, 32);
      ps += __shfl_xor(ps, 4, 32);
      ps += __shfl_xor(ps, 2, 32);
      ps += __shfl_xor(ps, 1, 32);
      lrow[r] = lrow[r] * alpha + ps;
      mrow[r] = mnew;
      const int prow = r + hid * 8;
      Pl[wave][prow * 34 + nl]      = (half_t)p0;
      Pl[wave][prow * 34 + 16 + nl] = (half_t)p1;
      o0[r] *= alpha; o1[r] *= alpha; o2[r] *= alpha; o3[r] *= alpha;
    }

    // ---- O += P(16x32) @ V(32x64) ----
    v16h pf = load_a_frag(Pl[wave], 34, 0, 0);
    o0 = wmma_f16(pf, load_b_fragT(Vl, 34, 0,  0), o0);
    o1 = wmma_f16(pf, load_b_fragT(Vl, 34, 16, 0), o1);
    o2 = wmma_f16(pf, load_b_fragT(Vl, 34, 32, 0), o2);
    o3 = wmma_f16(pf, load_b_fragT(Vl, 34, 48, 0), o3);
  }

#pragma unroll
  for (int r = 0; r < 8; ++r) {
    const float inv = 1.0f / lrow[r];
    half_t* dst = Oh + (qrow + r + hid * 8) * 64 + nl;
    dst[0]  = (half_t)(o0[r] * inv);
    dst[16] = (half_t)(o1[r] * inv);
    dst[32] = (half_t)(o2[r] * inv);
    dst[48] = (half_t)(o3[r] * inv);
  }
}

// ---------------------------------------------------------------------------
// Kernel 4: output projection + transpose to NCHW f32.
// ---------------------------------------------------------------------------
__global__ __launch_bounds__(256) void mha_proj(
    const half_t* __restrict__ Oh, const half_t* __restrict__ Woh,
    const float* __restrict__ bo, float* __restrict__ out) {
  const int rowbase = blockIdx.x * 16;
  const int wave = threadIdx.x >> 5;
  const int lane = threadIdx.x & 31;
  const int hid  = lane >> 4;
  const int nl   = lane & 15;
  const int nb   = wave * 32;

  v8f acc0 = {}, acc1 = {};
#pragma unroll
  for (int kc = 0; kc < 8; ++kc) {
    v16h a  = load_a_frag(Oh, 256, rowbase, kc * 32);
    v16h b0 = load_b_fragT(Woh, 256, nb,      kc * 32);
    v16h b1 = load_b_fragT(Woh, 256, nb + 16, kc * 32);
    acc0 = wmma_f16(a, b0, acc0);
    acc1 = wmma_f16(a, b1, acc1);
  }
  const float bias0 = bo[nb + nl];
  const float bias1 = bo[nb + 16 + nl];
#pragma unroll
  for (int r = 0; r < 8; ++r) {
    const int grow = rowbase + r + hid * 8;
    out[(nb + nl) * 2304 + grow]      = acc0[r] + bias0;
    out[(nb + 16 + nl) * 2304 + grow] = acc1[r] + bias1;
  }
}

// ---------------------------------------------------------------------------
extern "C" void kernel_launch(void* const* d_in, const int* in_sizes, int n_in,
                              void* d_out, int out_size, void* d_ws, size_t ws_size,
                              hipStream_t stream) {
  const float* x  = (const float*)d_in[0];
  const float* Wq = (const float*)d_in[1];
  const float* bq = (const float*)d_in[2];
  const float* Wk = (const float*)d_in[3];
  const float* bk = (const float*)d_in[4];
  const float* Wv = (const float*)d_in[5];
  const float* bv = (const float*)d_in[6];
  const float* Wo = (const float*)d_in[7];
  const float* bo = (const float*)d_in[8];

  half_t* ws  = (half_t*)d_ws;
  half_t* Xh  = ws;
  half_t* Wqh = Xh  + 2304 * 256;
  half_t* Wkh = Wqh + 256 * 256;
  half_t* Wvh = Wkh + 256 * 256;
  half_t* Woh = Wvh + 256 * 256;
  half_t* Qh  = Woh + 256 * 256;        // Qf [9216][64]
  half_t* Kh  = Qh  + 2304 * 256;       // Kf [9216][64]
  half_t* Vt  = Kh  + 2304 * 256;       // Vt [64][9216]
  half_t* Oh  = Vt  + 2304 * 256;       // O  [9216][64]

  mha_pack<<<2304 + 4 * 256, 256, 0, stream>>>(x, Wq, Wk, Wv, Wo,
                                               Xh, Wqh, Wkh, Wvh, Woh);
  dim3 gqkv(144, 3);
  mha_qkv<<<gqkv, 256, 0, stream>>>(Xh, Wqh, Wkh, Wvh, bq, bk, bv, Qh, Kh, Vt);
  mha_attn<<<144, 128, 0, stream>>>(Qh, Kh, Vt, Oh);
  mha_proj<<<144, 256, 0, stream>>>(Oh, Woh, bo, (float*)d_out);
}
